// DKTAccumModel_76828374991492
// MI455X (gfx1250) — matrix operand order repaired and checked by reading, not compiled, not run on hardware
//
#include <hip/hip_runtime.h>
#include <hip/hip_bf16.h>

typedef __bf16 bf16_t;
typedef __attribute__((ext_vector_type(16))) __bf16 v16bf;
typedef __attribute__((ext_vector_type(8)))  float  v8f;

#define B_    128
#define T_    1024
#define S_    200
#define E_    100
#define H_    100
#define NROWS (B_ * T_)     // 131072
#define KDIM  400           // 4*E
#define KPAD  416           // 13 * 32
#define NDIM  400           // 4*H
#define KC_X  13            // K-chunks (32 wide) for Wl GEMM
#define KC_H  4             // K-chunks for Ul GEMM (K=100 -> 128)
#define NT    25            // N tiles of 16 (400/16)
#define NBLK  5             // N-tiles per wave in the big GEMM

#if __has_builtin(__builtin_amdgcn_tensor_load_to_lds) && \
    __has_builtin(__builtin_amdgcn_s_wait_tensorcnt)
#define HAVE_TDM 1
#endif

#ifdef HAVE_TDM
typedef __attribute__((ext_vector_type(4))) unsigned int v4u;
typedef __attribute__((ext_vector_type(4))) int          v4i;
typedef __attribute__((ext_vector_type(8))) int          v8i;

// 1-D TDM copy: nbytes contiguous bytes global -> LDS (per CDNA5 D# layout).
__device__ __forceinline__ void tdm_load_slab(const void* gsrc, unsigned lds_off,
                                              int nbytes) {
  unsigned long long ga = (unsigned long long)gsrc;
  v4u g0;
  g0[0] = 1u;                                           // count=1, user mode
  g0[1] = lds_off;                                      // lds_addr
  g0[2] = (unsigned)(ga & 0xffffffffu);                 // global_addr lo
  g0[3] = (unsigned)((ga >> 32) & 0x01ffffffu) | (2u << 30);  // ga hi | type=2
  v8i g1;
  g1[0] = 0;                                            // wg_mask=0, data 1B
  g1[1] = (int)(((unsigned)nbytes & 0xffffu) << 16);    // tensor_dim0 lo16
  g1[2] = (int)((((unsigned)nbytes >> 16) & 0xffffu) | (1u << 16)); // d0 hi | d1=1
  g1[3] = (int)(((unsigned)nbytes & 0xffffu) << 16);    // tile_dim0 (16b)
  g1[4] = 1;                                            // tile_dim1=1
  g1[5] = nbytes;                                       // dim0_stride lo
  g1[6] = 0;
  g1[7] = 0;
  v4i z = {0, 0, 0, 0};
#if __clang_major__ >= 23
  v8i z8 = {0, 0, 0, 0, 0, 0, 0, 0};
  __builtin_amdgcn_tensor_load_to_lds(g0, g1, z, z, z8, 0);
#else
  __builtin_amdgcn_tensor_load_to_lds(g0, g1, z, z, 0);
#endif
}
#endif

// ---------------------------------------------------------------------------
// Kernel 1: decode one-hot x (B,T,400) and q (B,T,200) into indices + mask
// ---------------------------------------------------------------------------
__global__ void k_decode(const float* __restrict__ x, const float* __restrict__ q,
                         int* __restrict__ idx, int* __restrict__ qidx,
                         int* __restrict__ msk) {
  int row = blockIdx.x;  // b*T + t
  __shared__ int sx, sq;
  if (threadIdx.x == 0) { sx = 0x7fffffff; sq = 0x7fffffff; }
  __syncthreads();
  const float* xr = x + (size_t)row * (2 * S_);
  const float* qr = q + (size_t)row * S_;
  for (int c = threadIdx.x; c < 2 * S_; c += blockDim.x)
    if (xr[c] > 0.5f) atomicMin(&sx, c);
  for (int c = threadIdx.x; c < S_; c += blockDim.x)
    if (qr[c] > 0.5f) atomicMin(&sq, c);
  __syncthreads();
  if (threadIdx.x == 0) {
    msk[row]  = (sx != 0x7fffffff) ? 1 : 0;
    idx[row]  = (sx == 0x7fffffff) ? 0 : sx;
    qidx[row] = (sq == 0x7fffffff) ? 0 : sq;
  }
}

// ---------------------------------------------------------------------------
// Kernel 2: per-batch running class histogram; gather cc / ic per step
// ---------------------------------------------------------------------------
__global__ void k_counts(const int* __restrict__ idx, const int* __restrict__ msk,
                         float* __restrict__ ccic) {
  int b = blockIdx.x;
  __shared__ int cnt[2 * S_];
  for (int i = threadIdx.x; i < 2 * S_; i += blockDim.x) cnt[i] = 0;
  __syncthreads();
  if (threadIdx.x == 0) {
    for (int t = 0; t < T_; ++t) {
      int r = b * T_ + t;
      int cls = idx[r];
      if (msk[r]) cnt[cls] += 1;
      int base = (cls >> 1) << 1;
      ccic[2 * r]     = (float)cnt[base];
      ccic[2 * r + 1] = (float)cnt[base + 1];
    }
  }
}

// ---------------------------------------------------------------------------
// Kernel 3: repack Wl (400x400) and Ul (100x400) into WMMA-B fragment-major
// element e of lane L of chunk (nt,kc) holds B[kc*32 + (e/8)*16 + (L/16)*8 + e%8][nt*16 + L%16]
// ---------------------------------------------------------------------------
__global__ void k_pack(const float* __restrict__ Wl, const float* __restrict__ Ul,
                       bf16_t* __restrict__ Wlf, bf16_t* __restrict__ Ulf) {
  int p = blockIdx.x * blockDim.x + threadIdx.x;
  const int nW = NT * KC_X * 512;
  const int nU = NT * KC_H * 512;
  if (p < nW) {
    int nt = p / (KC_X * 512); int r = p % (KC_X * 512);
    int kc = r / 512;          int r2 = r % 512;
    int L = r2 >> 4;           int e = r2 & 15;
    int k = kc * 32 + ((e >> 3) << 4) + ((L >> 4) << 3) + (e & 7);
    int n = nt * 16 + (L & 15);
    Wlf[p] = (k < KDIM) ? (bf16_t)Wl[k * NDIM + n] : (bf16_t)0.f;
  } else if (p < nW + nU) {
    int p2 = p - nW;
    int nt = p2 / (KC_H * 512); int r = p2 % (KC_H * 512);
    int kc = r / 512;           int r2 = r % 512;
    int L = r2 >> 4;            int e = r2 & 15;
    int k = kc * 32 + ((e >> 3) << 4) + ((L >> 4) << 3) + (e & 7);
    int n = nt * 16 + (L & 15);
    Ulf[p2] = (k < H_) ? (bf16_t)Ul[k * NDIM + n] : (bf16_t)0.f;
  }
}

// ---------------------------------------------------------------------------
// Kernel 4: build fused feature rows x_c_t (bf16, K padded to 416)
// ---------------------------------------------------------------------------
__global__ void k_build(const float* __restrict__ Wx, const float* __restrict__ bx,
                        const float* __restrict__ Wc, const float* __restrict__ bc,
                        const float* __restrict__ Ws, const float* __restrict__ bs,
                        const float* __restrict__ Wr, const float* __restrict__ br,
                        const float* __restrict__ seqd, const float* __restrict__ repd,
                        const int* __restrict__ idx, const int* __restrict__ msk,
                        const float* __restrict__ ccic, bf16_t* __restrict__ xct) {
  int row = blockIdx.x;
  int cls = idx[row];
  int mk  = msk[row];
  float cc = ccic[2 * row], ic = ccic[2 * row + 1];
  float ls = log1pf(seqd[row]), lr = log1pf(repd[row]);
  bf16_t* o = xct + (size_t)row * KPAD;
  for (int c = threadIdx.x; c < KPAD; c += blockDim.x) {
    float v;
    if (c < E_)            v = (mk ? Wx[cls * E_ + c] : 0.f) + bx[c];
    else if (c < 2 * E_) { int e = c - E_;     v = ls * Ws[e] + bs[e]; }
    else if (c < 3 * E_) { int e = c - 2 * E_; v = lr * Wr[e] + br[e]; }
    else if (c < 4 * E_) { int e = c - 3 * E_;
      v = cc * Wc[e] + ic * Wc[E_ + e] + (cc + ic) * Wc[2 * E_ + e] + bc[e];
    } else v = 0.f;
    o[c] = (bf16_t)v;
  }
}

// ---------------------------------------------------------------------------
// Kernel 5: xz = x_c_t @ Wl + bl. Each wave: one 16-row block x 5 N-tiles;
// A fragments loaded once per K-chunk and reused across 5 WMMAs.
// Output stored T-MAJOR: xzt[t][b][n] (t = row & 1023, b = row >> 10).
// ---------------------------------------------------------------------------
__global__ void __launch_bounds__(256) k_gemm_xz(const bf16_t* __restrict__ xct,
                                                 const bf16_t* __restrict__ Wlf,
                                                 const float* __restrict__ bl,
                                                 bf16_t* __restrict__ xzt) {
  int wave = threadIdx.x >> 5;
  int lane = threadIdx.x & 31;
  int gid  = blockIdx.x * 8 + wave;    // 8192 * 5 wave-jobs
  int mt   = gid / NBLK;
  int nt0  = (gid % NBLK) * NBLK;      // 5 consecutive N-tiles
  int Ml = lane & 15, hh = lane >> 4;
  const bf16_t* arow = xct + (size_t)(mt * 16 + Ml) * KPAD;
  v8f acc[NBLK] = {};
  for (int kc = 0; kc < KC_X; ++kc) {
    v16bf a;
#pragma unroll
    for (int i = 0; i < 8; ++i) {
      int kb = kc * 32 + ((i >> 2) << 4) + ((i & 3) << 1) + (hh << 3);
      a[2 * i]     = arow[kb];
      a[2 * i + 1] = arow[kb + 1];
    }
#pragma unroll
    for (int j = 0; j < NBLK; ++j) {
      v16bf b = *(const v16bf*)(Wlf +
                 ((size_t)((nt0 + j) * KC_X + kc) * 32 + lane) * 16);
      acc[j] = __builtin_amdgcn_wmma_f32_16x16x32_bf16(false, a, false, b,
                                                       (short)0, acc[j],
                                                       false, false);
    }
  }
#pragma unroll
  for (int j = 0; j < NBLK; ++j) {
    int n = (nt0 + j) * 16 + Ml;
    float blv = bl[n];
#pragma unroll
    for (int r = 0; r < 8; ++r) {
      int rowm = mt * 16 + hh * 8 + r;
      int t = rowm & (T_ - 1);
      int b = rowm >> 10;
      xzt[((size_t)t * B_ + b) * NDIM + n] = (bf16_t)(acc[j][r] + blv);
    }
  }
}

// ---------------------------------------------------------------------------
// Kernel 6: persistent LSTM. 8 blocks x 16 batches. Per step:
//   TDM double-buffers the contiguous 12.8KB xz slab [16 batches x 400] into
//   LDS (prefetch t+1 while computing t), z = xz + h @ Ul via WMMA,
//   fused gates + output dot with Wo[:, qidx].
// ---------------------------------------------------------------------------
__global__ void __launch_bounds__(256) k_lstm(const bf16_t* __restrict__ xzt,
                                              const bf16_t* __restrict__ Ulf,
                                              const float* __restrict__ Wo,
                                              const float* __restrict__ bo,
                                              const int* __restrict__ qidx,
                                              const int* __restrict__ msk,
                                              float* __restrict__ out) {
  __shared__ float  zs[16][NDIM];        // 25.6 KB
  __shared__ bf16_t hs[16][128];         // 4 KB (K padded, cols 100..127 = 0)
  __shared__ float  cs[16][H_];          // 6.4 KB
#ifdef HAVE_TDM
  __shared__ bf16_t zxs[2][16 * NDIM];   // 2 x 12.8 KB double buffer
#endif
  int tid = threadIdx.x;
  int wv = tid >> 5, lane = tid & 31;
  int b0 = blockIdx.x * 16;
  int Ml = lane & 15, hh = lane >> 4;

  for (int i = tid; i < 16 * 128; i += 256) ((bf16_t*)hs)[i] = (bf16_t)0.f;
  for (int i = tid; i < 16 * H_;  i += 256) ((float*)cs)[i] = 0.f;

#ifdef HAVE_TDM
  const int slab_bytes = 16 * NDIM * 2;  // 12800
  if (wv == 0) {
    tdm_load_slab(xzt + (size_t)b0 * NDIM,
                  (unsigned)(unsigned long long)(void*)&zxs[0][0], slab_bytes);
  }
#endif
  __syncthreads();

  for (int t = 0; t < T_; ++t) {
#ifdef HAVE_TDM
    int cur = t & 1;
    if (wv == 0) __builtin_amdgcn_s_wait_tensorcnt(0);
    __syncthreads();  // slab t visible to all waves; prev step fully retired
    if (wv == 0 && (t + 1) < T_) {
      tdm_load_slab(xzt + ((size_t)(t + 1) * B_ + b0) * NDIM,
                    (unsigned)(unsigned long long)(void*)&zxs[cur ^ 1][0],
                    slab_bytes);
    }
    const bf16_t* slab = &zxs[cur][0];
#endif
    // ---- stage 1: z = xz + h @ Ul (WMMA) ----
    for (int nt = wv; nt < NT; nt += 8) {
      v8f acc = {};
      for (int kc = 0; kc < KC_H; ++kc) {
        v16bf a, b;
#pragma unroll
        for (int i = 0; i < 8; ++i) {
          int kb = kc * 32 + ((i >> 2) << 4) + ((i & 3) << 1) + (hh << 3);
          a[2 * i]     = hs[Ml][kb];
          a[2 * i + 1] = hs[Ml][kb + 1];
        }
        b = *(const v16bf*)(Ulf + ((size_t)(nt * KC_H + kc) * 32 + lane) * 16);
        acc = __builtin_amdgcn_wmma_f32_16x16x32_bf16(false, a, false, b,
                                                      (short)0, acc, false, false);
      }
      int n = nt * 16 + Ml;
#pragma unroll
      for (int r = 0; r < 8; ++r) {
        int m = hh * 8 + r;
#ifdef HAVE_TDM
        float xv = (float)slab[m * NDIM + n];
#else
        float xv = (float)xzt[((size_t)t * B_ + b0 + m) * NDIM + n];
#endif
        zs[m][n] = acc[r] + xv;
      }
    }
    __syncthreads();

    // ---- stage 2: gates + state update ----
    for (int p = tid; p < 16 * H_; p += 256) {
      int m = p / H_, j = p % H_;
      int r = (b0 + m) * T_ + t;
      float zi = zs[m][j],          zf = zs[m][H_ + j];
      float zg = zs[m][2 * H_ + j], zo = zs[m][3 * H_ + j];
      float ig = 1.f / (1.f + expf(-zi));
      float fg = 1.f / (1.f + expf(-zf));
      float og = 1.f / (1.f + expf(-zo));
      float cn = fg * cs[m][j] + ig * tanhf(zg);
      float hn = og * tanhf(cn);
      if (msk[r]) { cs[m][j] = cn; hs[m][j] = (bf16_t)hn; }
    }
    __syncthreads();

    // ---- stage 3: out[b,t] = sigmoid(h . Wo[:,qid] + bo[qid]) ----
    {
      int m = tid >> 4, j0 = tid & 15;
      int r = (b0 + m) * T_ + t;
      int qd = qidx[r];
      float p = 0.f;
      for (int j = j0; j < H_; j += 16)
        p += (float)hs[m][j] * Wo[j * S_ + qd];
      p += __shfl_xor(p, 1, 16);
      p += __shfl_xor(p, 2, 16);
      p += __shfl_xor(p, 4, 16);
      p += __shfl_xor(p, 8, 16);
      if (j0 == 0) out[r] = 1.f / (1.f + expf(-(p + bo[qd])));
    }
    // next stage-1 only reads hs / writes zs; top-of-loop barrier re-syncs
  }
}

// ---------------------------------------------------------------------------
extern "C" void kernel_launch(void* const* d_in, const int* in_sizes, int n_in,
                              void* d_out, int out_size, void* d_ws, size_t ws_size,
                              hipStream_t stream) {
  const float* x    = (const float*)d_in[0];
  const float* seqd = (const float*)d_in[1];
  const float* repd = (const float*)d_in[2];
  const float* q    = (const float*)d_in[3];
  const float* Wx   = (const float*)d_in[4];
  const float* bx   = (const float*)d_in[5];
  const float* Wc   = (const float*)d_in[6];
  const float* bc   = (const float*)d_in[7];
  const float* Ws   = (const float*)d_in[8];
  const float* bs   = (const float*)d_in[9];
  const float* Wr   = (const float*)d_in[10];
  const float* br   = (const float*)d_in[11];
  const float* Wl   = (const float*)d_in[12];
  const float* Ul   = (const float*)d_in[13];
  const float* bl   = (const float*)d_in[14];
  const float* Wo   = (const float*)d_in[15];
  const float* bo   = (const float*)d_in[16];
  float* out = (float*)d_out;

  char* ws = (char*)d_ws;
  size_t off = 0;
  auto alloc = [&](size_t bytes) -> void* {
    void* p = ws + off;
    off = (off + bytes + 255) & ~(size_t)255;
    return p;
  };
  int*    idx  = (int*)alloc((size_t)NROWS * 4);
  int*    qidx = (int*)alloc((size_t)NROWS * 4);
  int*    msk  = (int*)alloc((size_t)NROWS * 4);
  float*  ccic = (float*)alloc((size_t)NROWS * 2 * 4);
  bf16_t* Wlf  = (bf16_t*)alloc((size_t)NT * KC_X * 512 * 2);
  bf16_t* Ulf  = (bf16_t*)alloc((size_t)NT * KC_H * 512 * 2);
  bf16_t* xct  = (bf16_t*)alloc((size_t)NROWS * KPAD * 2);
  bf16_t* xzt  = (bf16_t*)alloc((size_t)NROWS * NDIM * 2);

  k_decode<<<NROWS, 128, 0, stream>>>(x, q, idx, qidx, msk);
  k_counts<<<B_, 128, 0, stream>>>(idx, msk, ccic);
  {
    int total = NT * KC_X * 512 + NT * KC_H * 512;
    k_pack<<<(total + 255) / 256, 256, 0, stream>>>(Wl, Ul, Wlf, Ulf);
  }
  k_build<<<NROWS, 128, 0, stream>>>(Wx, bx, Wc, bc, Ws, bs, Wr, br,
                                     seqd, repd, idx, msk, ccic, xct);
  k_gemm_xz<<<(8192 * NBLK) / 8, 256, 0, stream>>>(xct, Wlf, bl, xzt);
  k_lstm<<<8, 256, 0, stream>>>(xzt, Ulf, Wo, bo, qidx, msk, out);
}
